// N3Block_62878321214254
// MI455X (gfx1250) — matrix operand order repaired and checked by reading, not compiled, not run on hardware
//
#include <hip/hip_runtime.h>
#include <hip/hip_bf16.h>

// ---------------------------------------------------------------------------
// Non-local KNN block for MI455X (gfx1250, wave32, WMMA + TDM).
//
//   1) cvt_weights   : f32 -> f16 weights (w3 padded 8->16 OC rows)
//   2) conv3x3_wmma  : implicit-GEMM 3x3 conv. Full weight panel DMA'd into
//                      LDS once per block via tensor_load_to_lds (TDM,
//                      TENSORcnt), K-loop stages only the im2col A tile and
//                      issues v_wmma_f32_16x16x32_f16.
//   3) bn_stats      : per-channel mean/var -> fused scale/shift
//   4) bn_relu_cvt   : BN+ReLU, emit f16 activations for next WMMA conv
//   5) nonlocal_knn  : per 16x16 pixel tile: 441-offset 8-ch squared distance,
//                      separable 10x10 box sum in LDS, register top-7,
//                      softmax, 64-ch weighted gather, concat output.
// ---------------------------------------------------------------------------

typedef _Float16 h16;
typedef __attribute__((ext_vector_type(16))) _Float16 v16h;
typedef __attribute__((ext_vector_type(8)))  float    v8f;
typedef __attribute__((ext_vector_type(4)))  unsigned su4;
typedef __attribute__((ext_vector_type(8)))  unsigned su8;

#define BB   4
#define CC   64
#define HH   96
#define WW   96
#define EE   8
#define HWN  (HH*WW)        // 9216
#define RR   10
#define WSZ  21
#define NOFF (WSZ*WSZ)      // 441
#define KNN  7

// ---------------------------------------------------------------------------
// 1) weight conversion (w3 zero-padded to 16 output channels)
// ---------------------------------------------------------------------------
__global__ __launch_bounds__(256) void cvt_weights(
    const float* __restrict__ w1, const float* __restrict__ w2,
    const float* __restrict__ w3,
    h16* __restrict__ o1, h16* __restrict__ o2, h16* __restrict__ o3)
{
    int i = blockIdx.x * 256 + threadIdx.x;
    if (i < 64 * 576) { o1[i] = (h16)w1[i]; o2[i] = (h16)w2[i]; }
    if (i < 16 * 576) o3[i] = (i < 8 * 576) ? (h16)w3[i] : (h16)0.0f;
}

// ---------------------------------------------------------------------------
// 2) implicit-GEMM 3x3 SAME conv, NCHW.  M = B*H*W, N = OC(<=64), K = 576.
//    128 threads = 4 waves; block tile 64(M) x 16*NSUB(N); K step 32.
//    Weight panel [OCP][576] f16 DMA'd to LDS once per block by the TDM.
//    A tile [64][32] f16 staged per K-step (im2col with zero borders).
// ---------------------------------------------------------------------------
template<typename Tin, int NSUB>
__global__ __launch_bounds__(128) void conv3x3_wmma(
    const Tin* __restrict__ in, const h16* __restrict__ wts,
    float* __restrict__ out, const float* __restrict__ bias,
    int OCS /* channels actually stored */)
{
    constexpr int IC  = 64;
    constexpr int KT  = IC * 9;        // 576
    constexpr int OCP = 16 * NSUB;     // padded output channels in panel

    extern __shared__ char smem[];
    h16 (*As)[32] = (h16 (*)[32])smem;               // [64][32]  (4 KB)
    h16*  Wl      = (h16*)(smem + 64 * 32 * 2);      // [OCP][KT] (OCP*1152 B)

    const int tid   = threadIdx.x;
    const int wave  = tid >> 5;
    const int lane  = tid & 31;
    const int g     = lane >> 4;       // half-wave group
    const int lr    = lane & 15;
    const int mbase = blockIdx.x * 64;

    // --- one TDM load per block: weights [OCP][KT] f16, row stride KT ---
    if (wave == 0) {
        unsigned           lds_base = (unsigned)(size_t)Wl;   // low 32b = LDS offset
        unsigned long long ga       = (unsigned long long)(size_t)wts;
        su4 g0;
        g0[0] = 1u;                                           // count=1, user D#
        g0[1] = lds_base;                                     // lds_addr
        g0[2] = (unsigned)ga;                                 // global_addr[31:0]
        g0[3] = (unsigned)((ga >> 32) & 0x01FFFFFFull)        // global_addr[56:32]
              | (2u << 30);                                   // type = 2 (image)
        su8 g1;
        g1[0] = 1u << 16;                                     // data_size=1 (2 bytes)
        g1[1] = (unsigned)(KT & 0xFFFF) << 16;                // tensor_dim0 lo16
        g1[2] = (unsigned)(KT >> 16)                          // tensor_dim0 hi16
              | ((unsigned)(OCP & 0xFFFF) << 16);             // tensor_dim1 lo16
        g1[3] = (unsigned)(OCP >> 16)                         // tensor_dim1 hi16
              | ((unsigned)KT << 16);                         // tile_dim0 = KT
        g1[4] = (unsigned)OCP;                                // tile_dim1 (tile_dim2=0)
        g1[5] = (unsigned)KT;                                 // tensor_dim0_stride lo32
        g1[6] = 0u;                                           // stride0 hi / stride1 lo
        g1[7] = 0u;
        asm volatile("tensor_load_to_lds %0, %1" :: "s"(g0), "s"(g1) : "memory");
        __builtin_amdgcn_s_wait_tensorcnt(0);
    }
    __syncthreads();   // publish weight panel to all waves

    v8f acc[NSUB];
    const v8f vzero = { 0.f,0.f,0.f,0.f,0.f,0.f,0.f,0.f };
#pragma unroll
    for (int s = 0; s < NSUB; ++s) acc[s] = vzero;

    for (int kb = 0; kb < KT; kb += 32) {
        // --- stage A tile (im2col on the fly, zero-padded borders)
        for (int t = tid; t < 64 * 32; t += 128) {
            int ml = t >> 5, kk = t & 31;
            int m  = mbase + ml;
            int b  = m / HWN;  int rem = m - b * HWN;
            int hh = rem / WW; int ww  = rem - hh * WW;
            int k  = kb + kk;
            int ic = k / 9;    int r9  = k - ic * 9;
            int kh = r9 / 3;   int kw  = r9 - kh * 3;
            int ih = hh + kh - 1, iw = ww + kw - 1;
            h16 v = (h16)0.0f;
            if (ih >= 0 && ih < HH && iw >= 0 && iw < WW)
                v = (h16)in[((b * IC + ic) * HH + ih) * WW + iw];
            As[ml][kk] = v;
        }
        __syncthreads();

        // --- A fragment (ISA 16-bit A 16x32 layout)
        union { v16h v; unsigned u[8]; } af;
        const unsigned* arow = (const unsigned*)&As[wave * 16 + lr][0];
#pragma unroll
        for (int i = 0; i < 8; ++i) {
            int k = ((i >> 2) << 4) + ((i & 3) << 1) + (g << 3);
            af.u[i] = arow[k >> 1];
        }
#pragma unroll
        for (int s = 0; s < NSUB; ++s) {
            // --- B fragment straight from the persistent weight panel
            union { v16h v; unsigned u[8]; } bf;
            const unsigned* brow = (const unsigned*)(Wl + (s * 16 + lr) * KT);
#pragma unroll
            for (int i = 0; i < 8; ++i) {
                int k = kb + ((i >> 2) << 4) + ((i & 3) << 1) + (g << 3);
                bf.u[i] = brow[k >> 1];
            }
            acc[s] = __builtin_amdgcn_wmma_f32_16x16x32_f16(
                false, af.v, false, bf.v, (short)0, acc[s], false, false);
        }
        __syncthreads();   // protect As before next-iteration overwrite
    }

    // --- store (ISA f32 C/D layout: N = lane&15, M = g*8 + r)
#pragma unroll
    for (int s = 0; s < NSUB; ++s) {
        int n = s * 16 + lr;
        if (n < OCS) {
            float bv = bias ? bias[n] : 0.0f;
#pragma unroll
            for (int r = 0; r < 8; ++r) {
                int m  = mbase + wave * 16 + g * 8 + r;
                int b  = m / HWN;  int rem = m - b * HWN;
                int hh = rem / WW; int ww  = rem - hh * WW;
                out[((b * OCS + n) * HH + hh) * WW + ww] = acc[s][r] + bv;
            }
        }
    }
}

// ---------------------------------------------------------------------------
// 3) per-channel BN stats -> fused scale/shift  (st[c], st[64+c])
// ---------------------------------------------------------------------------
__global__ __launch_bounds__(256) void bn_stats(
    const float* __restrict__ x, const float* __restrict__ gamma,
    const float* __restrict__ beta, float* __restrict__ st)
{
    const int c = blockIdx.x;
    __shared__ float s1[256], s2[256];
    float sum = 0.f, sq = 0.f;
    for (int i = threadIdx.x; i < BB * HWN; i += 256) {
        int b = i / HWN; int p = i - b * HWN;
        float v = x[(b * CC + c) * HWN + p];
        sum += v; sq += v * v;
    }
    s1[threadIdx.x] = sum; s2[threadIdx.x] = sq;
    __syncthreads();
    for (int off = 128; off > 0; off >>= 1) {
        if (threadIdx.x < off) {
            s1[threadIdx.x] += s1[threadIdx.x + off];
            s2[threadIdx.x] += s2[threadIdx.x + off];
        }
        __syncthreads();
    }
    if (threadIdx.x == 0) {
        const float inv_n = 1.0f / (float)(BB * HWN);
        float mean = s1[0] * inv_n;
        float var  = s2[0] * inv_n - mean * mean;
        float sc   = gamma[c] * rsqrtf(var + 1e-5f);
        st[c]      = sc;
        st[64 + c] = beta[c] - mean * sc;
    }
}

// ---------------------------------------------------------------------------
// 4) BN + ReLU, emit f16
// ---------------------------------------------------------------------------
__global__ __launch_bounds__(256) void bn_relu_cvt(
    const float* __restrict__ x, const float* __restrict__ st,
    h16* __restrict__ y, int total)
{
    int i = blockIdx.x * 256 + threadIdx.x;
    if (i >= total) return;
    int c = (i / HWN) & 63;
    float v = x[i] * st[c] + st[64 + c];
    y[i] = (h16)fmaxf(v, 0.0f);
}

// ---------------------------------------------------------------------------
// 5) non-local KNN stage.  One 256-thread block per 16x16 pixel tile.
// ---------------------------------------------------------------------------
__global__ __launch_bounds__(256) void nonlocal_knn(
    const float* __restrict__ xe, const float* __restrict__ x,
    float* __restrict__ out)
{
    __shared__ h16   xs[EE][48][48];   // 36 KB, f16 embedding tile + halo
    __shared__ float sp[25][26];       // squared-distance plane
    __shared__ float rs[25][16];       // row box sums

    const int bb = blockIdx.z;
    const int h0 = blockIdx.y * 16;
    const int w0 = blockIdx.x * 16;
    const int tid = threadIdx.x;
    const int ty = tid >> 4, tx = tid & 15;
    const int h = h0 + ty, w = w0 + tx;

    // load embedding region [h0-14, h0+33] x [w0-14, w0+33], zero outside
    for (int t = tid; t < EE * 48 * 48; t += 256) {
        int ch = t / (48 * 48); int r = t - ch * 48 * 48;
        int hh = r / 48;        int ww = r - hh * 48;
        int gh = h0 - 14 + hh,  gw = w0 - 14 + ww;
        float v = 0.f;
        if (gh >= 0 && gh < HH && gw >= 0 && gw < WW)
            v = xe[((bb * EE + ch) * HH + gh) * WW + gw];
        xs[ch][hh][ww] = (h16)v;
    }
    __syncthreads();

    float dk[KNN]; int ik[KNN];
#pragma unroll
    for (int k = 0; k < KNN; ++k) { dk[k] = 3.0e38f; ik[k] = 0; }

    for (int o = 0; o < NOFF; ++o) {
        const int dy = o / WSZ - RR;
        const int dx = o - (o / WSZ) * WSZ - RR;

        // s plane at h' = h0-4+sh, w' = w0-4+sw  (zero outside image)
        for (int t = tid; t < 25 * 25; t += 256) {
            int sh = t / 25, sw = t - sh * 25;
            int gh = h0 - 4 + sh, gw = w0 - 4 + sw;
            float s = 0.f;
            if (gh >= 0 && gh < HH && gw >= 0 && gw < WW) {
                int ra = sh + 10, ca = sw + 10;
                int rb = ra + dy, cb = ca + dx;
#pragma unroll
                for (int ch = 0; ch < EE; ++ch) {
                    float d = (float)xs[ch][ra][ca] - (float)xs[ch][rb][cb];
                    s += d * d;
                }
            }
            sp[sh][sw] = s;
        }
        __syncthreads();

        // row sums over w' in [w-4, w+5]
        for (int t = tid; t < 25 * 16; t += 256) {
            int sh = t >> 4, cx = t & 15;
            float a = 0.f;
#pragma unroll
            for (int j = 0; j < 10; ++j) a += sp[sh][cx + j];
            rs[sh][cx] = a;
        }
        __syncthreads();

        // column sums over h' in [h-4, h+5] -> D at this thread's pixel
        float D = 0.f;
#pragma unroll
        for (int j = 0; j < 10; ++j) D += rs[ty + j][tx];

        const int nh = h + dy, nw = w + dx;
        const bool valid = (nh >= 0) && (nh < HH) && (nw >= 0) && (nw < WW)
                           && !((dy == 0) && (dx == 0));
        if (valid && D < dk[KNN - 1]) {
            int k = KNN - 1;
            while (k > 0 && dk[k - 1] > D) {
                dk[k] = dk[k - 1]; ik[k] = ik[k - 1]; --k;
            }
            dk[k] = D; ik[k] = o;
        }
        // next iteration's sp-write barrier orders against rs reads
    }

    // softmax over -D (dk sorted ascending -> dk[0] is max of -D)
    float wk[KNN]; int ofs[KNN];
    float wsum = 0.f;
#pragma unroll
    for (int k = 0; k < KNN; ++k) { wk[k] = __expf(dk[0] - dk[k]); wsum += wk[k]; }
    const float inv = 1.0f / wsum;
#pragma unroll
    for (int k = 0; k < KNN; ++k) {
        wk[k] *= inv;
        int q = ik[k];
        int dy = q / WSZ - RR, dx = q - (q / WSZ) * WSZ - RR;
        ofs[k] = (h + dy) * WW + (w + dx);
    }

    const float* xb = x   + (size_t)bb * CC * HWN;
    float*       ob = out + (size_t)bb * (2 * CC) * HWN;
    const int    pix = h * WW + w;
    for (int c = 0; c < CC; ++c) {
        const float* xc = xb + c * HWN;
        float acc = 0.f;
#pragma unroll
        for (int k = 0; k < KNN; ++k) acc += wk[k] * xc[ofs[k]];
        ob[c * HWN + pix]        = xc[pix];   // passthrough x
        ob[(CC + c) * HWN + pix] = acc;       // aggregated neighbors
    }
}

// ---------------------------------------------------------------------------
// launcher
// ---------------------------------------------------------------------------
extern "C" void kernel_launch(void* const* d_in, const int* in_sizes, int n_in,
                              void* d_out, int out_size, void* d_ws, size_t ws_size,
                              hipStream_t stream)
{
    const float* x  = (const float*)d_in[0];
    // d_in[1] = flows (unused by reference math)
    const float* w1 = (const float*)d_in[2];
    const float* g1 = (const float*)d_in[3];
    const float* b1 = (const float*)d_in[4];
    const float* w2 = (const float*)d_in[5];
    const float* g2 = (const float*)d_in[6];
    const float* b2 = (const float*)d_in[7];
    const float* w3 = (const float*)d_in[8];
    const float* b3 = (const float*)d_in[9];
    float* out = (float*)d_out;

    char* ws = (char*)d_ws;
    // workspace layout (bytes)
    h16*   w1h   = (h16*)  (ws + 0);                  //  73,728
    h16*   w2h   = (h16*)  (ws + 73728);              //  73,728
    h16*   w3h   = (h16*)  (ws + 147456);             //  18,432
    float* h1raw = (float*)(ws + 165888);             // 9,437,184
    h16*   h1a   = (h16*)  (ws + 9603072);            // 4,718,592
    float* h2raw = (float*)(ws + 14321664);           // 9,437,184
    h16*   h2a   = (h16*)  (ws + 23758848);           // 4,718,592
    float* xebuf = (float*)(ws + 28477440);           // 1,179,648
    float* st1   = (float*)(ws + 29657088);           //       512
    float* st2   = (float*)(ws + 29657600);           //       512

    const int MBLK  = (BB * HWN) / 64;    // 576 GEMM M-tiles
    const int TOTAL = BB * CC * HWN;      // 2,359,296 activations

    const size_t SH64 = 64 * 32 * 2 + 64 * 576 * 2;   // A tile + 64-ch panel
    const size_t SH16 = 64 * 32 * 2 + 16 * 576 * 2;   // A tile + 16-ch panel

    cvt_weights<<<144, 256, 0, stream>>>(w1, w2, w3, w1h, w2h, w3h);

    conv3x3_wmma<float, 4><<<MBLK, 128, SH64, stream>>>(x, w1h, h1raw, nullptr, 64);
    bn_stats<<<64, 256, 0, stream>>>(h1raw, g1, b1, st1);
    bn_relu_cvt<<<(TOTAL + 255) / 256, 256, 0, stream>>>(h1raw, st1, h1a, TOTAL);

    conv3x3_wmma<h16, 4><<<MBLK, 128, SH64, stream>>>(h1a, w2h, h2raw, nullptr, 64);
    bn_stats<<<64, 256, 0, stream>>>(h2raw, g2, b2, st2);
    bn_relu_cvt<<<(TOTAL + 255) / 256, 256, 0, stream>>>(h2raw, st2, h2a, TOTAL);

    // 64 -> 8 channels, OC padded to 16 for WMMA; bias b3 folded into store
    conv3x3_wmma<h16, 1><<<MBLK, 128, SH16, stream>>>(h2a, w3h, xebuf, b3, 8);

    dim3 grid(WW / 16, HH / 16, BB);     // (6, 6, 4)
    nonlocal_knn<<<grid, 256, 0, stream>>>(xebuf, x, out);
}
